// RSSM_72112500900139
// MI455X (gfx1250) — matrix operand order repaired and checked by reading, not compiled, not run on hardware
//
#include <hip/hip_runtime.h>
#include <hip/hip_bf16.h>
#include <stdint.h>

typedef unsigned short u16;
typedef __attribute__((ext_vector_type(16))) __bf16 v16bf;
typedef __attribute__((ext_vector_type(8)))  float  v8f;

union FragU { unsigned int u[8]; v16bf v; };

__device__ __forceinline__ u16 f2bf(float f) {
    unsigned int u = __builtin_bit_cast(unsigned int, f);
    unsigned int r = u + 0x7FFFu + ((u >> 16) & 1u);
    return (u16)(r >> 16);
}

static constexpr int Bsz = 2048;
static constexpr int Tsteps = 50;
static constexpr int H = 1024;

// ---------- CDNA5 async global->LDS copies (ASYNCcnt-tracked, bypass VGPRs) ----------
__device__ __forceinline__ void async_b128(unsigned lds_addr, const u16* gaddr) {
    asm volatile("global_load_async_to_lds_b128 %0, %1, off"
                 :: "v"(lds_addr), "v"(gaddr) : "memory");
}
__device__ __forceinline__ void async_b64(unsigned lds_addr, const u16* gaddr) {
    asm volatile("global_load_async_to_lds_b64 %0, %1, off"
                 :: "v"(lds_addr), "v"(gaddr) : "memory");
}
__device__ __forceinline__ void wait_async() {
    asm volatile("s_wait_asynccnt 0x0" ::: "memory");
}
// generic LDS pointer -> 32-bit LDS byte offset (flat LDS addr == low 32 bits)
__device__ __forceinline__ unsigned lds_off(const void* p) {
    return (unsigned)(uintptr_t)p;
}

// ---------- fragment loaders (ISA 7.12.2 16-bit layouts, wave32) ----------
// A (16x32, M x K): lane L: m = L%16, half = L/16; VGPR j: K = (j/4)*16 + half*8 + (j%4)*2
__device__ __forceinline__ v16bf load_frag_a(const u16* s, int mBase, int lane) {
    const int m = mBase + (lane & 15);
    const int hf = lane >> 4;
    FragU f;
#pragma unroll
    for (int j = 0; j < 8; ++j) {
        const int k = (j >> 2) * 16 + hf * 8 + (j & 3) * 2;
        f.u[j] = *(const unsigned int*)&s[m * 40 + k];
    }
    return f.v;
}
// B (32x16, K x N) from W rows (N x K): lane L: n = L%16, half = L/16; VGPR j: K = (j/4)*16 + half*8 + (j%4)*2
__device__ __forceinline__ v16bf load_frag_b(const u16* s, int nBase, int lane) {
    const int n = nBase + (lane & 15);
    const int hf = lane >> 4;
    FragU f;
#pragma unroll
    for (int j = 0; j < 8; ++j) {
        const int k = (j >> 2) * 16 + hf * 8 + (j & 3) * 2;
        f.u[j] = *(const unsigned int*)&s[n * 40 + k];
    }
    return f.v;
}

// ---------- generic GEMM: out[b,n] = act( sum_k A[b,k]*W[n,k] + bias[n] ) ----------
template <bool RELU, bool OUTBF>
__global__ __launch_bounds__(256) void gemm_wmma(
    const u16* __restrict__ A, int lda, int K,
    const u16* __restrict__ W,
    const float* __restrict__ bias,
    void* __restrict__ outp, int ldo)
{
    constexpr int BM = 128, BN = 64, BK = 32, STR = 40;
    __shared__ __align__(16) u16 As[2 * BM * STR];
    __shared__ __align__(16) u16 Ws[2 * BN * STR];
    const int tid = threadIdx.x;
    const int wave = tid >> 5, lane = tid & 31;
    const int wm = wave >> 1, wn = wave & 1;      // 4x2 waves -> 32x32 each
    const int rowBase = blockIdx.y * BM;
    const int colBase = blockIdx.x * BN;
    const unsigned AsAddr = lds_off(As);
    const unsigned WsAddr = lds_off(Ws);

    // A tile: 512 x (8 elem) chunks -> 2 per thread; W tile: 256 chunks -> 1 per thread
    const int ra0 = tid >> 2, ca0 = (tid & 3) * 8;
    const int ra1 = (tid + 256) >> 2, ca1 = ((tid + 256) & 3) * 8;
    const int rb = tid >> 2, cb = (tid & 3) * 8;

    auto issue = [&](int k0, int buf) {
        const unsigned ab = AsAddr + (unsigned)(buf * BM * STR * 2);
        const unsigned wb = WsAddr + (unsigned)(buf * BN * STR * 2);
        async_b128(ab + (unsigned)(ra0 * STR + ca0) * 2,
                   A + (size_t)(rowBase + ra0) * lda + k0 + ca0);
        async_b128(ab + (unsigned)(ra1 * STR + ca1) * 2,
                   A + (size_t)(rowBase + ra1) * lda + k0 + ca1);
        async_b128(wb + (unsigned)(rb * STR + cb) * 2,
                   W + (size_t)(colBase + rb) * K + k0 + cb);
    };

    v8f acc[2][2] = {};
    issue(0, 0);

    int it = 0;
    for (int k0 = 0; k0 < K; k0 += BK, ++it) {
        const int buf = it & 1;
        wait_async();
        __syncthreads();
        if (k0 + BK < K) issue(k0 + BK, buf ^ 1);   // prefetch next tile
        const u16* Asb = As + buf * BM * STR;
        const u16* Wsb = Ws + buf * BN * STR;
        const v16bf a0 = load_frag_a(Asb, wm * 32, lane);
        const v16bf a1 = load_frag_a(Asb, wm * 32 + 16, lane);
        const v16bf b0 = load_frag_b(Wsb, wn * 32, lane);
        const v16bf b1 = load_frag_b(Wsb, wn * 32 + 16, lane);
        acc[0][0] = __builtin_amdgcn_wmma_f32_16x16x32_bf16(false, a0, false, b0, (short)0, acc[0][0], false, false);
        acc[0][1] = __builtin_amdgcn_wmma_f32_16x16x32_bf16(false, a0, false, b1, (short)0, acc[0][1], false, false);
        acc[1][0] = __builtin_amdgcn_wmma_f32_16x16x32_bf16(false, a1, false, b0, (short)0, acc[1][0], false, false);
        acc[1][1] = __builtin_amdgcn_wmma_f32_16x16x32_bf16(false, a1, false, b1, (short)0, acc[1][1], false, false);
    }

    const int hf = lane >> 4, ln = lane & 15;
#pragma unroll
    for (int tm = 0; tm < 2; ++tm)
#pragma unroll
        for (int tn = 0; tn < 2; ++tn) {
            const int col = colBase + wn * 32 + tn * 16 + ln;
            const float bv = bias[col];
#pragma unroll
            for (int j = 0; j < 8; ++j) {
                const int row = rowBase + wm * 32 + tm * 16 + j + 8 * hf;
                float v = acc[tm][tn][j] + bv;
                if (RELU) v = fmaxf(v, 0.f);
                if (OUTBF) ((u16*)outp)[(size_t)row * ldo + col] = f2bf(v);
                else       ((float*)outp)[(size_t)row * ldo + col] = v;
            }
        }
}

// ---------- fused GRU: b1 = (1-z)*tanh(inn + r*hn) + z*h ----------
// Xg = [rnn_in ∥ belief] bf16 [Bsz x 2048]; Wg = [[W_ih ∥ W_hh] rows r,z,n] bf16 [3072 x 2048]
__global__ __launch_bounds__(256) void gru_wmma(
    const u16* __restrict__ Xg, const u16* __restrict__ Wg,
    const float* __restrict__ b_ih, const float* __restrict__ b_hh,
    float* __restrict__ state_b, float* __restrict__ out_beliefs_t,
    u16* __restrict__ Xq)
{
    constexpr int BM = 128, BN = 32, BK = 32, STR = 40, KT = 2048;
    __shared__ __align__(16) u16 As[2 * BM * STR];
    __shared__ __align__(16) u16 WsR[2 * BN * STR];
    __shared__ __align__(16) u16 WsZ[2 * BN * STR];
    __shared__ __align__(16) u16 WsN[2 * BN * STR];
    const int tid = threadIdx.x;
    const int wave = tid >> 5, lane = tid & 31;
    const int wm = wave >> 1, wn = wave & 1;      // 4x2 waves -> 32x16 each
    const int rowBase = blockIdx.y * BM;
    const int colBase = blockIdx.x * BN;
    const unsigned AsAddr = lds_off(As);
    const unsigned RAddr = lds_off(WsR), ZAddr = lds_off(WsZ), NAddr = lds_off(WsN);

    const int ra0 = tid >> 2, ca0 = (tid & 3) * 8;
    const int ra1 = (tid + 256) >> 2, ca1 = ((tid + 256) & 3) * 8;
    const int rb = tid >> 3, cb = (tid & 7) * 4;   // gate tiles: b64 chunks, all 256 lanes

    auto issue = [&](int k0, int buf) {
        const unsigned ab = AsAddr + (unsigned)(buf * BM * STR * 2);
        const unsigned bo = (unsigned)(buf * BN * STR * 2);
        async_b128(ab + (unsigned)(ra0 * STR + ca0) * 2,
                   Xg + (size_t)(rowBase + ra0) * KT + k0 + ca0);
        async_b128(ab + (unsigned)(ra1 * STR + ca1) * 2,
                   Xg + (size_t)(rowBase + ra1) * KT + k0 + ca1);
        const size_t wrow = (size_t)(colBase + rb) * KT + k0 + cb;
        const unsigned loff = (unsigned)(rb * STR + cb) * 2;
        async_b64(RAddr + bo + loff, Wg + wrow);
        async_b64(ZAddr + bo + loff, Wg + wrow + (size_t)H * KT);
        async_b64(NAddr + bo + loff, Wg + wrow + (size_t)2 * H * KT);
    };

    v8f accR[2] = {}, accZ[2] = {}, accI[2] = {}, accH[2] = {};
    issue(0, 0);

    int it = 0;
    for (int k0 = 0; k0 < KT; k0 += BK, ++it) {
        const int buf = it & 1;
        wait_async();
        __syncthreads();
        if (k0 + BK < KT) issue(k0 + BK, buf ^ 1);
        const u16* Asb = As + buf * BM * STR;
        const v16bf a0 = load_frag_a(Asb, wm * 32, lane);
        const v16bf a1 = load_frag_a(Asb, wm * 32 + 16, lane);
        const v16bf bR = load_frag_b(WsR + buf * BN * STR, wn * 16, lane);
        const v16bf bZ = load_frag_b(WsZ + buf * BN * STR, wn * 16, lane);
        const v16bf bN = load_frag_b(WsN + buf * BN * STR, wn * 16, lane);
        accR[0] = __builtin_amdgcn_wmma_f32_16x16x32_bf16(false, a0, false, bR, (short)0, accR[0], false, false);
        accR[1] = __builtin_amdgcn_wmma_f32_16x16x32_bf16(false, a1, false, bR, (short)0, accR[1], false, false);
        accZ[0] = __builtin_amdgcn_wmma_f32_16x16x32_bf16(false, a0, false, bZ, (short)0, accZ[0], false, false);
        accZ[1] = __builtin_amdgcn_wmma_f32_16x16x32_bf16(false, a1, false, bZ, (short)0, accZ[1], false, false);
        if (k0 < H) {  // uniform branch: first K half feeds inn, second feeds hn
            accI[0] = __builtin_amdgcn_wmma_f32_16x16x32_bf16(false, a0, false, bN, (short)0, accI[0], false, false);
            accI[1] = __builtin_amdgcn_wmma_f32_16x16x32_bf16(false, a1, false, bN, (short)0, accI[1], false, false);
        } else {
            accH[0] = __builtin_amdgcn_wmma_f32_16x16x32_bf16(false, a0, false, bN, (short)0, accH[0], false, false);
            accH[1] = __builtin_amdgcn_wmma_f32_16x16x32_bf16(false, a1, false, bN, (short)0, accH[1], false, false);
        }
    }

    const int hf = lane >> 4, ln = lane & 15;
    const int col = colBase + wn * 16 + ln;
    const float br = b_ih[col] + b_hh[col];
    const float bz = b_ih[H + col] + b_hh[H + col];
    const float bi = b_ih[2 * H + col];
    const float bh = b_hh[2 * H + col];
#pragma unroll
    for (int tm = 0; tm < 2; ++tm)
#pragma unroll
        for (int j = 0; j < 8; ++j) {
            const int row = rowBase + wm * 32 + tm * 16 + j + 8 * hf;
            const float rg = 1.f / (1.f + expf(-(accR[tm][j] + br)));
            const float zg = 1.f / (1.f + expf(-(accZ[tm][j] + bz)));
            const float nv = tanhf(accI[tm][j] + bi + rg * (accH[tm][j] + bh));
            const float h  = state_b[(size_t)row * H + col];
            const float b1 = (1.f - zg) * nv + zg * h;
            state_b[(size_t)row * H + col] = b1;
            out_beliefs_t[(size_t)row * H + col] = b1;
            Xq[(size_t)row * 2048 + col] = f2bf(b1);
        }
}

// ---------- elementwise helpers ----------
__global__ void convert_bf16(u16* __restrict__ dst, const float* __restrict__ src, int n) {
    const int i = blockIdx.x * 256 + threadIdx.x;
    if (i < n) dst[i] = f2bf(src[i]);
}

__global__ void build_wg(u16* __restrict__ Wg, const float* __restrict__ W_ih,
                         const float* __restrict__ W_hh) {
    const size_t i = (size_t)blockIdx.x * 256 + threadIdx.x;
    if (i >= (size_t)3072 * 2048) return;
    const size_t n = i >> 11;
    const int k = (int)(i & 2047);
    const float v = (k < 1024) ? W_ih[n * 1024 + k] : W_hh[n * 1024 + (k - 1024)];
    Wg[i] = f2bf(v);
}

// stage per-step bf16 inputs: Xsa=[post_state∥a_t], Xg[:,1024:]=belief, Xq[:,1024:]=o_t
__global__ void prep_step(const float* __restrict__ state_s, const float* __restrict__ act_t,
                          const float* __restrict__ state_b, const float* __restrict__ obs_t,
                          u16* __restrict__ Xsa, u16* __restrict__ Xg, u16* __restrict__ Xq) {
    const int i = blockIdx.x * 256 + threadIdx.x;
    const int n1 = Bsz * 288, n2 = n1 + Bsz * 1024, n3 = n2 + Bsz * 1024;
    if (i < n1) {
        const int b = i / 288, c = i % 288;
        const float v = (c < 256) ? state_s[b * 256 + c] : act_t[b * 32 + (c - 256)];
        Xsa[b * 288 + c] = f2bf(v);
    } else if (i < n2) {
        const int j = i - n1; const int b = j >> 10, c = j & 1023;
        Xg[(size_t)b * 2048 + 1024 + c] = f2bf(state_b[j]);
    } else if (i < n3) {
        const int j = i - n2; const int b = j >> 10, c = j & 1023;
        Xq[(size_t)b * 2048 + 1024 + c] = f2bf(obs_t[j]);
    }
}

// stats [B x 512] -> mean/std/sample [B x 256] (+ optional state update)
__global__ void sample_kernel(const float* __restrict__ stats, const float* __restrict__ noise_t,
                              float* __restrict__ om, float* __restrict__ osd,
                              float* __restrict__ osmp, float* __restrict__ state_q) {
    const int i = blockIdx.x * 256 + threadIdx.x;
    if (i >= Bsz * 256) return;
    const int b = i >> 8, s = i & 255;
    const float m  = stats[b * 512 + s];
    const float lg = stats[b * 512 + 256 + s];
    const float x  = expf(lg);
    const float sp = (x > 20.f) ? x : log1pf(expf(x));
    const float sd = sp + 0.1f;
    const float smp = m + sd * noise_t[i];
    om[i] = m; osd[i] = sd; osmp[i] = smp;
    if (state_q) state_q[i] = smp;
}

// ---------- host ----------
extern "C" void kernel_launch(void* const* d_in, const int* in_sizes, int n_in,
                              void* d_out, int out_size, void* d_ws, size_t ws_size,
                              hipStream_t stream) {
    const float* prev_state   = (const float*)d_in[0];
    const float* actions      = (const float*)d_in[1];
    const float* prev_belief  = (const float*)d_in[2];
    const float* observations = (const float*)d_in[3];
    const float* prior_noise  = (const float*)d_in[4];
    const float* post_noise   = (const float*)d_in[5];
    const float* W_sa = (const float*)d_in[6];
    const float* b_sa = (const float*)d_in[7];
    const float* W_ih = (const float*)d_in[8];
    const float* W_hh = (const float*)d_in[9];
    const float* b_ih = (const float*)d_in[10];
    const float* b_hh = (const float*)d_in[11];
    const float* W_pb = (const float*)d_in[12];
    const float* b_pb = (const float*)d_in[13];
    const float* W_ps = (const float*)d_in[14];
    const float* b_ps = (const float*)d_in[15];
    const float* W_qb = (const float*)d_in[16];
    const float* b_qb = (const float*)d_in[17];
    const float* W_qs = (const float*)d_in[18];
    const float* b_qs = (const float*)d_in[19];

    float* out = (float*)d_out;
    char* ws = (char*)d_ws;
    size_t off = 0;
    auto alloc = [&](size_t bytes) -> void* {
        void* p = ws + off; off += (bytes + 255) & ~(size_t)255; return p;
    };
    u16* Wg_bf  = (u16*)alloc((size_t)3072 * 2048 * 2);
    u16* Wsa_bf = (u16*)alloc((size_t)1024 * 288 * 2);
    u16* Wpb_bf = (u16*)alloc((size_t)1024 * 1024 * 2);
    u16* Wps_bf = (u16*)alloc((size_t)512 * 1024 * 2);
    u16* Wqb_bf = (u16*)alloc((size_t)1024 * 2048 * 2);
    u16* Wqs_bf = (u16*)alloc((size_t)512 * 1024 * 2);
    u16* Xsa    = (u16*)alloc((size_t)Bsz * 288 * 2);
    u16* Xg     = (u16*)alloc((size_t)Bsz * 2048 * 2);
    u16* Xq     = (u16*)alloc((size_t)Bsz * 2048 * 2);
    u16* Hid    = (u16*)alloc((size_t)Bsz * 1024 * 2);
    float* Stats   = (float*)alloc((size_t)Bsz * 512 * 4);
    float* state_s = (float*)alloc((size_t)Bsz * 256 * 4);
    float* state_b = (float*)alloc((size_t)Bsz * 1024 * 4);

    const size_t TB = (size_t)Tsteps * Bsz;
    float* out_beliefs = out;
    float* out_ps   = out + TB * 1024;
    float* out_pm   = out_ps + TB * 256;
    float* out_pstd = out_pm + TB * 256;
    float* out_qs   = out_pstd + TB * 256;
    float* out_qm   = out_qs + TB * 256;
    float* out_qstd = out_qm + TB * 256;

    auto g1 = [](size_t n) { return dim3((unsigned)((n + 255) / 256)); };

    // one-time weight prep (per launch; deterministic)
    convert_bf16<<<g1(1024 * 288), 256, 0, stream>>>(Wsa_bf, W_sa, 1024 * 288);
    convert_bf16<<<g1(1024 * 1024), 256, 0, stream>>>(Wpb_bf, W_pb, 1024 * 1024);
    convert_bf16<<<g1(512 * 1024), 256, 0, stream>>>(Wps_bf, W_ps, 512 * 1024);
    convert_bf16<<<g1(1024 * 2048), 256, 0, stream>>>(Wqb_bf, W_qb, 1024 * 2048);
    convert_bf16<<<g1(512 * 1024), 256, 0, stream>>>(Wqs_bf, W_qs, 512 * 1024);
    build_wg<<<g1((size_t)3072 * 2048), 256, 0, stream>>>(Wg_bf, W_ih, W_hh);
    hipMemcpyAsync(state_s, prev_state, (size_t)Bsz * 256 * 4, hipMemcpyDeviceToDevice, stream);
    hipMemcpyAsync(state_b, prev_belief, (size_t)Bsz * 1024 * 4, hipMemcpyDeviceToDevice, stream);

    const dim3 blk(256);
    const dim3 gSA(1024 / 64, Bsz / 128);
    const dim3 gGRU(1024 / 32, Bsz / 128);
    const dim3 gH(1024 / 64, Bsz / 128);
    const dim3 gS(512 / 64, Bsz / 128);
    const size_t prepN = (size_t)Bsz * (288 + 1024 + 1024);

    for (int t = 0; t < Tsteps; ++t) {
        const float* act_t = actions + (size_t)t * Bsz * 32;
        const float* obs_t = observations + (size_t)t * Bsz * 1024;
        const float* pn_t  = prior_noise + (size_t)t * Bsz * 256;
        const float* qn_t  = post_noise + (size_t)t * Bsz * 256;
        const size_t oB = (size_t)t * Bsz * 1024, oS = (size_t)t * Bsz * 256;

        prep_step<<<g1(prepN), blk, 0, stream>>>(state_s, act_t, state_b, obs_t, Xsa, Xg, Xq);
        // rnn_in = relu([s,a] @ W_sa^T + b) -> Xg[:, 0:1024] (bf16)
        gemm_wmma<true, true><<<gSA, blk, 0, stream>>>(Xsa, 288, 288, Wsa_bf, b_sa, Xg, 2048);
        // GRU -> state_b (f32), beliefs out, Xq[:, 0:1024] (bf16)
        gru_wmma<<<gGRU, blk, 0, stream>>>(Xg, Wg_bf, b_ih, b_hh, state_b, out_beliefs + oB, Xq);
        // prior
        gemm_wmma<true, true><<<gH, blk, 0, stream>>>(Xq, 2048, 1024, Wpb_bf, b_pb, Hid, 1024);
        gemm_wmma<false, false><<<gS, blk, 0, stream>>>(Hid, 1024, 1024, Wps_bf, b_ps, Stats, 512);
        sample_kernel<<<g1((size_t)Bsz * 256), blk, 0, stream>>>(
            Stats, pn_t, out_pm + oS, out_pstd + oS, out_ps + oS, nullptr);
        // posterior
        gemm_wmma<true, true><<<gH, blk, 0, stream>>>(Xq, 2048, 2048, Wqb_bf, b_qb, Hid, 1024);
        gemm_wmma<false, false><<<gS, blk, 0, stream>>>(Hid, 1024, 1024, Wqs_bf, b_qs, Stats, 512);
        sample_kernel<<<g1((size_t)Bsz * 256), blk, 0, stream>>>(
            Stats, qn_t, out_qm + oS, out_qstd + oS, out_qs + oS, state_s);
    }
}
